// Binning_41549513622107
// MI455X (gfx1250) — matrix operand order repaired and checked by reading, not compiled, not run on hardware
//
#include <hip/hip_runtime.h>

typedef __attribute__((ext_vector_type(16))) __bf16 v16bf;
typedef __attribute__((ext_vector_type(16))) unsigned short v16u;
typedef __attribute__((ext_vector_type(8)))  float v8f;

__device__ __forceinline__ unsigned short f2bf(float f) {
  unsigned int u = __float_as_uint(f);
  unsigned int r = u + 0x7FFFu + ((u >> 16) & 1u);
  return (unsigned short)(r >> 16);
}
__device__ __forceinline__ float bf2f(unsigned short h) {
  return __uint_as_float(((unsigned int)h) << 16);
}

// ---------------------------------------------------------------------------
// Weight packing: OIHW f32 -> bf16 WMMA A-fragments
// layout: [tap(9)][coTile][chunk][lane(32)][elem(16)]
// A 16x32 packing (contiguous K, matches B loads): lanes 0-15: K in {0..15}; lanes 16-31: K in {16..31}
// ---------------------------------------------------------------------------
__global__ void k_pack_weights(const float* __restrict__ w, unsigned short* __restrict__ wp,
                               int cout, int cin, int coTiles, int nChunks) {
  size_t total = (size_t)9 * coTiles * nChunks * 512;
  size_t idx = (size_t)blockIdx.x * blockDim.x + threadIdx.x;
  if (idx >= total) return;
  int e    = (int)(idx & 15);
  int lane = (int)((idx >> 4) & 31);
  size_t t = idx >> 9;
  int chunk = (int)(t % nChunks); t /= nChunks;
  int coT   = (int)(t % coTiles); t /= coTiles;
  int tap   = (int)t;
  int m  = lane & 15;
  int hl = lane >> 4;
  int kk = hl * 16 + e;
  int co = coT * 16 + m;
  int ci = chunk * 32 + kk;
  float v = 0.0f;
  if (co < cout && ci < cin)
    v = w[((size_t)co * cin + ci) * 9 + tap];
  wp[idx] = f2bf(v);
}

// ---------------------------------------------------------------------------
// Input packing: concat(global_features, features[:,sp]) -> bf16 HWC, Cp=96
// ---------------------------------------------------------------------------
__global__ void k_pack_input(const float* __restrict__ feats, const float* __restrict__ gf,
                             unsigned short* __restrict__ out, int sp) {
  int idx = blockIdx.x * blockDim.x + threadIdx.x;   // (b, y*160+x)
  if (idx >= 2 * 160 * 160) return;
  int b = idx / 25600;
  int p = idx - b * 25600;
  unsigned short* o = out + (size_t)idx * 96;
  for (int c = 0; c < 3; ++c) o[c] = f2bf(gf[b * 3 + c]);
  const float* f = feats + ((size_t)(b * 4 + sp) * 64) * 25600 + p;
  for (int c = 0; c < 64; ++c) o[3 + c] = f2bf(f[(size_t)c * 25600]);
  for (int c = 67; c < 96; ++c) o[c] = 0;
}

// ---------------------------------------------------------------------------
// Implicit-GEMM 3x3 conv, bf16 in/out (HWC), f32 WMMA accumulate.
// Block = 256 threads = 8 waves; wave w -> cout tile [16w,16w+16).
// Block covers NT*16 output pixels in one row; each wave keeps NT independent
// accumulator chains sharing one A fragment (amortizes A loads, hides XDL
// RAW latency). K-loop: 9 taps (rolled) x NCHUNK unrolled 32-chunks.
// PAD=0 (VALID) has no bounds predication at all (provably in-bounds).
// ---------------------------------------------------------------------------
template<int PAD, int NCHUNK, int NT>
__global__ __launch_bounds__(256) void k_conv3x3_wmma(
    const unsigned short* __restrict__ in, unsigned short* __restrict__ out,
    const unsigned short* __restrict__ wp, const float* __restrict__ bias,
    int inWH, int Cp, int outWH, int CpOut, int cout, int coTiles, int leaky)
{
  const int wave = (int)(threadIdx.x >> 5);
  if (wave >= coTiles) return;
  const int lane = (int)(threadIdx.x & 31);
  const int n  = lane & 15;     // B column / output pixel within a 16-tile
  const int hi = lane >> 4;     // lane half
  const int xb = blockIdx.x * (16 * NT);
  const int y  = blockIdx.y;
  const int b  = blockIdx.z;
  const unsigned short* inB = in + (size_t)b * inWH * inWH * Cp;

  int px[NT], pxc[NT];
#pragma unroll
  for (int t = 0; t < NT; ++t) {
    px[t]  = xb + t * 16 + n;
    pxc[t] = (px[t] < outWH) ? px[t] : (outWH - 1);   // clamp addr; never stored
  }

  v8f acc[NT] = {};

#pragma unroll 1
  for (int tap = 0; tap < 9; ++tap) {
    const int ky = tap / 3, kx = tap - ky * 3;
    const int iy = y + ky - PAD;
    const unsigned short* ws =
        wp + ((size_t)(tap * coTiles + wave) * NCHUNK * 32 + lane) * 16;
    const unsigned short* src[NT];
    bool ok[NT];
#pragma unroll
    for (int t = 0; t < NT; ++t) {
      const int ix = pxc[t] + kx - PAD;
      if (PAD) {
        ok[t] = (iy >= 0) && (iy < inWH) && (ix >= 0) && (ix < inWH);
        src[t] = inB + ((size_t)(ok[t] ? iy : 0) * inWH + (ok[t] ? ix : 0)) * Cp + hi * 16;
      } else {
        ok[t] = true;
        src[t] = inB + ((size_t)iy * inWH + ix) * Cp + hi * 16;
      }
    }
#pragma unroll
    for (int c = 0; c < NCHUNK; ++c) {
      v16u au = *(const v16u*)(ws + (size_t)c * 512);   // packed A fragment
#pragma unroll
      for (int t = 0; t < NT; ++t) {
        v16u bu;
        if (PAD) {
          bu = (v16u){};
          if (ok[t]) bu = *(const v16u*)(src[t] + c * 32);
        } else {
          bu = *(const v16u*)(src[t] + c * 32);         // K = chunk*32 + hi*16 + [0..15]
        }
        acc[t] = __builtin_amdgcn_wmma_f32_16x16x32_bf16(
            false, __builtin_bit_cast(v16bf, au),
            false, __builtin_bit_cast(v16bf, bu),
            (short)0, acc[t], false, false);
      }
    }
  }

#pragma unroll
  for (int t = 0; t < NT; ++t) {
    if (px[t] < outWH) {
      unsigned short* dst = out + (((size_t)b * outWH + y) * outWH + px[t]) * CpOut;
#pragma unroll
      for (int r = 0; r < 8; ++r) {
        int co = wave * 16 + hi * 8 + r;                // D layout: M = r + 8*hi
        if (co < cout) {
          float v = acc[t][r] + bias[co];
          if (leaky) v = (v > 0.0f) ? v : v * 0.01f;
          dst[co] = f2bf(v);
        }
      }
    }
  }
}

// ---------------------------------------------------------------------------
// concat(state, emb) along channels -> 256-ch HWC buffer
// ---------------------------------------------------------------------------
__global__ void k_concat(const unsigned short* __restrict__ a, const unsigned short* __restrict__ b,
                         unsigned short* __restrict__ cat, int npix) {
  int p = blockIdx.x * blockDim.x + threadIdx.x;
  if (p >= npix) return;
  const unsigned short* pa = a + (size_t)p * 128;
  const unsigned short* pb = b + (size_t)p * 128;
  unsigned short* pc = cat + (size_t)p * 256;
  for (int c = 0; c < 128; ++c) { pc[c] = pa[c]; pc[128 + c] = pb[c]; }
}

// ---------------------------------------------------------------------------
// softmax(logits 4ch) * cropped radiance, accumulate /spp into radcls [b][12][140][140]
// ---------------------------------------------------------------------------
__global__ void k_wrad_accum(const unsigned short* __restrict__ logits,
                             const float* __restrict__ radiance,
                             float* __restrict__ radcls, int sp, int isFirst) {
  int idx = blockIdx.x * blockDim.x + threadIdx.x;   // (b, y*140+x)
  if (idx >= 2 * 140 * 140) return;
  int b = idx / 19600;
  int p = idx - b * 19600;
  int y = p / 140, x = p - y * 140;
  const unsigned short* lg = logits + (size_t)idx * 4;
  float l[4], m = -1e30f;
  for (int c = 0; c < 4; ++c) { l[c] = bf2f(lg[c]); m = fmaxf(m, l[c]); }
  float s = 0.0f;
  for (int c = 0; c < 4; ++c) { l[c] = __expf(l[c] - m); s += l[c]; }
  float inv = 1.0f / s;
  float r[3];
  for (int j = 0; j < 3; ++j)
    r[j] = radiance[((size_t)(b * 4 + sp) * 3 + j) * 25600 + (size_t)(y + 10) * 160 + (x + 10)];
  float* o = radcls + (size_t)b * 12 * 19600 + p;
  for (int c = 0; c < 4; ++c)
    for (int j = 0; j < 3; ++j) {
      float v = l[c] * inv * r[j] * 0.25f;
      size_t ofs = (size_t)(c * 3 + j) * 19600;
      if (isFirst) o[ofs] = v; else o[ofs] += v;
    }
}

// ---------------------------------------------------------------------------
// per-class 25-way softmax over predicted kernels: bf16 HWC(100) -> f32 CHW
// ---------------------------------------------------------------------------
__global__ void k_kernel_softmax(const unsigned short* __restrict__ kin, float* __restrict__ ksm) {
  int idx = blockIdx.x * blockDim.x + threadIdx.x;   // (b, pixel, class)
  if (idx >= 2 * 20164 * 4) return;
  int c = idx & 3;
  int t = idx >> 2;
  int b = t / 20164;
  int p = t - b * 20164;
  const unsigned short* src = kin + ((size_t)b * 20164 + p) * 100 + c * 25;
  float v[25], m = -1e30f;
  for (int k = 0; k < 25; ++k) { v[k] = bf2f(src[k]); m = fmaxf(m, v[k]); }
  float s = 0.0f;
  for (int k = 0; k < 25; ++k) { v[k] = __expf(v[k] - m); s += v[k]; }
  float inv = 1.0f / s;
  float* dst = ksm + ((size_t)b * 100 + c * 25) * 20164 + p;
  for (int k = 0; k < 25; ++k) dst[(size_t)k * 20164] = v[k] * inv;
}

// ---------------------------------------------------------------------------
// final kernel application: out[b,j,136,136] = sum_c sum_{dy,dx} ker * data
// (kernel cropped by 3 on each side; data window slides over 140x140)
// ---------------------------------------------------------------------------
__global__ void k_apply(const float* __restrict__ ksm, const float* __restrict__ radcls,
                        float* __restrict__ out) {
  int idx = blockIdx.x * blockDim.x + threadIdx.x;
  if (idx >= 2 * 3 * 18496) return;
  int b = idx / (3 * 18496);
  int r0 = idx - b * 3 * 18496;
  int j = r0 / 18496;
  int p = r0 - j * 18496;
  int yy = p / 136, xx = p - yy * 136;
  float acc = 0.0f;
  for (int c = 0; c < 4; ++c) {
    const float* K = ksm + ((size_t)b * 100 + c * 25) * 20164 + (size_t)(yy + 3) * 142 + (xx + 3);
    const float* D = radcls + ((size_t)b * 12 + c * 3 + j) * 19600;
    for (int dy = 0; dy < 5; ++dy)
      for (int dx = 0; dx < 5; ++dx)
        acc += K[(size_t)(dy * 5 + dx) * 20164] * D[(size_t)(yy + dy) * 140 + (xx + dx)];
  }
  out[idx] = acc;
}

// ---------------------------------------------------------------------------
// host helpers
// ---------------------------------------------------------------------------
template<int PAD, int NCHUNK>
static void launch_conv(hipStream_t s, const unsigned short* in, unsigned short* out,
                        const unsigned short* wp, const float* bias,
                        int inWH, int Cp, int outWH, int CpOut,
                        int cout, int coTiles, int leaky) {
  constexpr int NT = 2;
  dim3 g((outWH + 16 * NT - 1) / (16 * NT), outWH, 2);
  k_conv3x3_wmma<PAD, NCHUNK, NT><<<g, 256, 0, s>>>(
      in, out, wp, bias, inWH, Cp, outWH, CpOut, cout, coTiles, leaky);
}

static void launch_packw(hipStream_t s, const float* w, unsigned short* wp,
                         int cout, int cin, int coTiles, int nChunks) {
  size_t total = (size_t)9 * coTiles * nChunks * 512;
  k_pack_weights<<<(unsigned)((total + 255) / 256), 256, 0, s>>>(w, wp, cout, cin, coTiles, nChunks);
}

extern "C" void kernel_launch(void* const* d_in, const int* in_sizes, int n_in,
                              void* d_out, int out_size, void* d_ws, size_t ws_size,
                              hipStream_t stream) {
  (void)in_sizes; (void)n_in; (void)out_size;
  const float* feats    = (const float*)d_in[0];   // (2,4,64,160,160)
  const float* radiance = (const float*)d_in[1];   // (2,4,3,160,160)
  const float* gfeat    = (const float*)d_in[2];   // (2,3,1,1)
  const float* emb_w[7]; const float* emb_b[7];
  for (int i = 0; i < 7; ++i) { emb_w[i] = (const float*)d_in[3 + 2 * i]; emb_b[i] = (const float*)d_in[4 + 2 * i]; }
  const float* cls_w[3]; const float* cls_b[3];
  for (int i = 0; i < 3; ++i) { cls_w[i] = (const float*)d_in[17 + 2 * i]; cls_b[i] = (const float*)d_in[18 + 2 * i]; }
  const float* rec_w0 = (const float*)d_in[23]; const float* rec_b0 = (const float*)d_in[24];
  const float* rec_w1 = (const float*)d_in[25]; const float* rec_b1 = (const float*)d_in[26];
  const float* kp_w0  = (const float*)d_in[27]; const float* kp_b0  = (const float*)d_in[28];
  const float* kp_w1  = (const float*)d_in[29]; const float* kp_b1  = (const float*)d_in[30];

  char* ws = (char*)d_ws; size_t off = 0;
  auto carve = [&](size_t bytes) -> void* {
    void* p = ws + off; off = (off + bytes + 255) & ~(size_t)255; return p;
  };
  unsigned short* pw_emb[7];
  pw_emb[0] = (unsigned short*)carve((size_t)9 * 8 * 3 * 512 * 2);
  for (int i = 1; i < 7; ++i) pw_emb[i] = (unsigned short*)carve((size_t)9 * 8 * 4 * 512 * 2);
  unsigned short* pw_cls0 = (unsigned short*)carve((size_t)9 * 8 * 4 * 512 * 2);
  unsigned short* pw_cls1 = (unsigned short*)carve((size_t)9 * 8 * 4 * 512 * 2);
  unsigned short* pw_cls2 = (unsigned short*)carve((size_t)9 * 1 * 4 * 512 * 2);
  unsigned short* pw_rec0 = (unsigned short*)carve((size_t)9 * 8 * 8 * 512 * 2);
  unsigned short* pw_rec1 = (unsigned short*)carve((size_t)9 * 8 * 4 * 512 * 2);
  unsigned short* pw_kp0  = (unsigned short*)carve((size_t)9 * 8 * 4 * 512 * 2);
  unsigned short* pw_kp1  = (unsigned short*)carve((size_t)9 * 7 * 4 * 512 * 2);
  unsigned short* bufA    = (unsigned short*)carve((size_t)160 * 160 * 128 * 2 * 2);
  unsigned short* bufB    = (unsigned short*)carve((size_t)160 * 160 * 128 * 2 * 2);
  unsigned short* bufEmb  = (unsigned short*)carve((size_t)146 * 146 * 128 * 2 * 2);
  unsigned short* bufSt   = (unsigned short*)carve((size_t)146 * 146 * 128 * 2 * 2);
  unsigned short* bufCat  = (unsigned short*)carve((size_t)146 * 146 * 256 * 2 * 2);
  float* radcls = (float*)carve((size_t)2 * 12 * 140 * 140 * 4);
  float* ksm    = (float*)carve((size_t)2 * 100 * 142 * 142 * 4);
  if (off > ws_size) return;   // insufficient scratch

  // ---- pack all weights (cheap, deterministic each call) ----
  launch_packw(stream, emb_w[0], pw_emb[0], 128, 67, 8, 3);
  for (int i = 1; i < 7; ++i) launch_packw(stream, emb_w[i], pw_emb[i], 128, 128, 8, 4);
  launch_packw(stream, cls_w[0], pw_cls0, 128, 128, 8, 4);
  launch_packw(stream, cls_w[1], pw_cls1, 128, 128, 8, 4);
  launch_packw(stream, cls_w[2], pw_cls2,   4, 128, 1, 4);
  launch_packw(stream, rec_w0,   pw_rec0, 128, 256, 8, 8);
  launch_packw(stream, rec_w1,   pw_rec1, 128, 128, 8, 4);
  launch_packw(stream, kp_w0,    pw_kp0,  128, 128, 8, 4);
  launch_packw(stream, kp_w1,    pw_kp1,  100, 128, 7, 4);

  for (int sp = 0; sp < 4; ++sp) {
    // concat(gf, features[:,sp]) -> bufA (Cp=96, 160x160)
    k_pack_input<<<(2 * 160 * 160 + 255) / 256, 256, 0, stream>>>(feats, gfeat, bufA, sp);

    // embedding chain (VALID, all leaky): 160 -> 146
    launch_conv<0, 3>(stream, bufA, bufB, pw_emb[0], emb_b[0], 160,  96, 158, 128, 128, 8, 1);
    launch_conv<0, 4>(stream, bufB, bufA, pw_emb[1], emb_b[1], 158, 128, 156, 128, 128, 8, 1);
    launch_conv<0, 4>(stream, bufA, bufB, pw_emb[2], emb_b[2], 156, 128, 154, 128, 128, 8, 1);
    launch_conv<0, 4>(stream, bufB, bufA, pw_emb[3], emb_b[3], 154, 128, 152, 128, 128, 8, 1);
    launch_conv<0, 4>(stream, bufA, bufB, pw_emb[4], emb_b[4], 152, 128, 150, 128, 128, 8, 1);
    launch_conv<0, 4>(stream, bufB, bufA, pw_emb[5], emb_b[5], 150, 128, 148, 128, 128, 8, 1);
    launch_conv<0, 4>(stream, bufA, bufEmb, pw_emb[6], emb_b[6], 148, 128, 146, 128, 128, 8, 1);

    // classifier (VALID, last linear): 146 -> 140, cout=4
    launch_conv<0, 4>(stream, bufEmb, bufA, pw_cls0, cls_b[0], 146, 128, 144, 128, 128, 8, 1);
    launch_conv<0, 4>(stream, bufA, bufB, pw_cls1, cls_b[1], 144, 128, 142, 128, 128, 8, 1);
    launch_conv<0, 4>(stream, bufB, bufA, pw_cls2, cls_b[2], 142, 128, 140,   4,   4, 1, 0);

    // softmax + weighted radiance accumulation
    k_wrad_accum<<<(2 * 140 * 140 + 255) / 256, 256, 0, stream>>>(
        bufA, radiance, radcls, sp, sp == 0 ? 1 : 0);

    if (sp == 0) {
      hipMemcpyAsync(bufSt, bufEmb, (size_t)146 * 146 * 128 * 2 * 2,
                     hipMemcpyDeviceToDevice, stream);
    } else {
      // recurrent (SAME): concat(state, emb) -> 256ch -> 128 -> 128(linear)
      k_concat<<<(2 * 146 * 146 + 255) / 256, 256, 0, stream>>>(bufSt, bufEmb, bufCat, 2 * 146 * 146);
      launch_conv<1, 8>(stream, bufCat, bufB, pw_rec0, rec_b0, 146, 256, 146, 128, 128, 8, 1);
      launch_conv<1, 4>(stream, bufB, bufSt, pw_rec1, rec_b1, 146, 128, 146, 128, 128, 8, 0);
    }
  }

  // kernel predictor (VALID, last linear): 146 -> 142, cout=100
  launch_conv<0, 4>(stream, bufSt, bufA, pw_kp0, kp_b0, 146, 128, 144, 128, 128, 8, 1);
  launch_conv<0, 4>(stream, bufA, bufB, pw_kp1, kp_b1, 144, 128, 142, 100, 100, 7, 0);

  // per-class 25-way softmax, then apply cropped 5x5 kernels
  k_kernel_softmax<<<(2 * 20164 * 4 + 255) / 256, 256, 0, stream>>>(bufB, ksm);
  k_apply<<<(2 * 3 * 18496 + 255) / 256, 256, 0, stream>>>(ksm, radcls, (float*)d_out);
}
